// Bi_LSTM_CRF_6442450944304
// MI455X (gfx1250) — matrix operand order repaired and checked by reading, not compiled, not run on hardware
//
#include <hip/hip_runtime.h>
#include <cstdint>

// ---------------------------------------------------------------------------
// Bi-LSTM-CRF for MI455X (gfx1250): wave32, v_wmma_f32_16x16x32_f16,
// Tensor Data Mover (tensor_load_to_lds + s_wait_tensorcnt) for x_t staging.
// E=128, H=256, V=30552, T=9, B=64, L=512
// ---------------------------------------------------------------------------

typedef __attribute__((ext_vector_type(16))) _Float16 v16h;
typedef __attribute__((ext_vector_type(8)))  _Float16 v8h;
typedef __attribute__((ext_vector_type(8)))  float    v8f;
typedef __attribute__((ext_vector_type(4)))  unsigned int u32x4;
typedef __attribute__((ext_vector_type(8)))  int i32x8;
typedef __attribute__((ext_vector_type(4)))  int i32x4;

#define LL 512
#define BB 64
#define EE 128
#define HH 256
#define TT 9

#ifndef __has_builtin
#define __has_builtin(x) 0
#endif
#if defined(__HIP_DEVICE_COMPILE__) && \
    __has_builtin(__builtin_amdgcn_tensor_load_to_lds) && \
    __has_builtin(__builtin_amdgcn_s_wait_tensorcnt)
#define USE_TDM 1
#endif

// Load a 16x32 f16 WMMA A-fragment from a row-major f16 matrix.
// p must already point at: base + (tile_row + (lane&15))*ld + kc*32 + (lane>>4)*8
// Layout (ISA 7.12.2): halves 0-7  -> K = kc*32 + sel*8 + 0..7
//                      halves 8-15 -> K = kc*32 + 16 + sel*8 + 0..7
__device__ __forceinline__ v16h load_a_frag(const _Float16* p) {
  union { v16h v; v8h h[2]; } u;
  u.h[0] = *(const v8h*)(p);
  u.h[1] = *(const v8h*)(p + 16);
  return u.v;
}

#ifdef USE_TDM
// Issue a TDM DMA: contiguous 2-D tile (BB rows x kin f16) from global memory
// into LDS at byte offset lds_byte_off. D# per CDNA5 ISA 8.3/8.4:
//   group0: count=1, lds_addr, global_addr[56:0], type=2
//   group1: data_size=1 (2B), tensor_dim0=kin, tensor_dim1=BB,
//           tile_dim0=kin, tile_dim1=BB, dim0_stride=kin, dim1_stride=kin*BB
__device__ __forceinline__ void tdm_issue_x(const _Float16* gsrc,
                                            unsigned lds_byte_off, int kin) {
  unsigned long long ga = (unsigned long long)gsrc;
  u32x4 g0;
  g0.x = 1u;                                              // count=1, user mode
  g0.y = lds_byte_off;                                    // lds_addr
  g0.z = (unsigned)ga;                                    // global_addr[31:0]
  g0.w = (unsigned)((ga >> 32) & 0x1FFFFFFull) | (2u << 30); // [56:32] | type=2
  unsigned td0 = (unsigned)kin;
  unsigned td1 = (unsigned)BB;
  unsigned long long st1 = (unsigned long long)kin * (unsigned long long)BB;
  i32x8 g1;
  g1[0] = (int)(1u << 16);                                // data_size = 2 bytes
  g1[1] = (int)((td0 & 0xFFFFu) << 16);                   // tensor_dim0[15:0]
  g1[2] = (int)((td0 >> 16) | ((td1 & 0xFFFFu) << 16));   // dim0 hi | dim1 lo
  g1[3] = (int)((td1 >> 16) | ((td0 & 0xFFFFu) << 16));   // dim1 hi | tile_dim0
  g1[4] = (int)(td1 & 0xFFFFu);                           // tile_dim1 | tile_dim2=0
  g1[5] = (int)td0;                                       // dim0_stride[31:0]
  g1[6] = (int)((st1 & 0xFFFFull) << 16);                 // stride0 hi=0 | stride1 lo
  g1[7] = (int)(st1 >> 16);                               // stride1[47:16]
  i32x4 z4 = {};
#if __clang_major__ >= 23
  i32x8 z8 = {};
  __builtin_amdgcn_tensor_load_to_lds(g0, g1, z4, z4, z8, 0);
#else
  __builtin_amdgcn_tensor_load_to_lds(g0, g1, z4, z4, 0);
#endif
}
#endif // USE_TDM

// ---------------------------------------------------------------------------
// Pack W (row-major f32, rows n = output, cols k) into the WMMA-B f16 layout:
//   out[((nt*KC + kc)*32 + lane)*16 + h] = W[n][k],
//   n = nt*16 + (lane&15), k = kc*32 + (h>>3)*16 + (lane>>4)*8 + (h&7)
// A-operand K-space is the concat [Wa cols (Ka) | Wb cols (Kb)].
// Rows n >= Nvalid are zero-padded (classifier: 9 -> 16).
// ---------------------------------------------------------------------------
__global__ void pack_weights(const float* __restrict__ Wa, int Ka,
                             const float* __restrict__ Wb, int Kb,
                             int Nvalid, int KC,
                             _Float16* __restrict__ out, int total) {
  int idx = blockIdx.x * blockDim.x + threadIdx.x;
  if (idx >= total) return;
  int h    = idx & 15;
  int lane = (idx >> 4) & 31;
  int rest = idx >> 9;
  int kc   = rest % KC;
  int nt   = rest / KC;
  int n = nt * 16 + (lane & 15);
  int k = kc * 32 + (h >> 3) * 16 + (lane >> 4) * 8 + (h & 7);
  float v = 0.f;
  if (n < Nvalid) {
    if (k < Ka) v = Wa[(size_t)n * Ka + k];
    else        v = Wb[(size_t)n * Kb + (k - Ka)];
  }
  out[idx] = (_Float16)v;
}

// ---------------------------------------------------------------------------
// Embedding gather + f32->f16: x[t][b][k] = emb[ids[b][t]][k]
// ---------------------------------------------------------------------------
__global__ void embed_gather(const int* __restrict__ ids,
                             const float* __restrict__ emb,
                             _Float16* __restrict__ xf16) {
  int idx = blockIdx.x * blockDim.x + threadIdx.x; // L*B*(E/4) = 1,048,576
  int k4 = idx & 31;
  int b  = (idx >> 5) & 63;
  int t  = idx >> 11;
  if (t >= LL) return;
  int tok = ids[b * LL + t];
  float4 v = *(const float4*)(emb + (size_t)tok * EE + k4 * 4);
  _Float16* o = xf16 + ((size_t)t * BB + b) * EE + k4 * 4;
  o[0] = (_Float16)v.x; o[1] = (_Float16)v.y;
  o[2] = (_Float16)v.z; o[3] = (_Float16)v.w;
}

// ---------------------------------------------------------------------------
// Persistent bidirectional LSTM layer. Grid = 2 blocks (dir 0 fwd, 1 bwd),
// 1024 threads = 32 waves per block. Each step:
//   z = [h_{t-1} | x_t] @ Wcat^T + bias   (K = 256 + KIN, N = 1024)
// h lives in LDS; x_t is double-buffered in LDS, staged by the Tensor Data
// Mover one step ahead (overlapped with the WMMA chain); c lives in VGPRs.
// Wave w owns combos c=2w,2w+1; combo c -> (m = c&3, j = c>>2); its 4 tiles
// are the SAME 16 hidden columns in gates i/f/g/o, so the nonlinearity and
// cell update are register-local to the wave. 2 barriers per step (TDM path).
// ---------------------------------------------------------------------------
template <int KIN>
__global__ __launch_bounds__(1024)
void lstm_bidir(const _Float16* __restrict__ xin,      // (L, B, KIN) f16
                const _Float16* __restrict__ wpack_f,  // packed (1024 x (256+KIN))
                const _Float16* __restrict__ wpack_b,
                const float* __restrict__ bias_f,      // 4H = 1024
                const float* __restrict__ bias_b,
                _Float16* __restrict__ hout) {         // (L, B, 512) f16
  constexpr int KC  = (HH + KIN) / 32;
  constexpr int KCH = HH / 32; // 8 chunks of h

  extern __shared__ _Float16 smem[];
  _Float16* Hls  = smem;                               // BB*HH halves (32 KB)
  _Float16* Xls0 = smem + BB * HH;                     // BB*KIN halves
  _Float16* Xls1 = smem + BB * HH + BB * KIN;          // BB*KIN halves

  const int dir = blockIdx.x;
  const _Float16* wpack = dir ? wpack_b : wpack_f;
  const float*    bias  = dir ? bias_b  : bias_f;
  const int col_off = dir * HH;

  const int tid  = threadIdx.x;
  const int lane = tid & 31;
  const int wave = tid >> 5;
  const int n15  = lane & 15;
  const int sel  = lane >> 4;

  const int c0 = 2 * wave, c1 = 2 * wave + 1;
  const int mm[2] = { c0 & 3, c1 & 3 };
  const int jj[2] = { c0 >> 2, c1 >> 2 };

  // zero initial h
  for (int i = tid; i < BB * HH / 8; i += 1024) {
    v8h z = {};
    *(v8h*)(Hls + i * 8) = z;
  }

  float cst[2][8];
#pragma unroll
  for (int c = 0; c < 2; ++c)
#pragma unroll
    for (int i = 0; i < 8; ++i) cst[c][i] = 0.f;

  float bval[2][4];
#pragma unroll
  for (int c = 0; c < 2; ++c)
#pragma unroll
    for (int g = 0; g < 4; ++g) bval[c][g] = bias[g * HH + jj[c] * 16 + n15];

#ifdef USE_TDM
  { // prologue: stage x_{t0} into buffer 0 (TDM, drained before first barrier)
    const int t0 = dir ? (LL - 1) : 0;
    if (wave == 0) {
      tdm_issue_x(xin + (size_t)t0 * BB * KIN,
                  (unsigned)(BB * HH * sizeof(_Float16)), KIN);
      __builtin_amdgcn_s_wait_tensorcnt(0);
    }
  }
#endif

  for (int it = 0; it < LL; ++it) {
    const int t = dir ? (LL - 1 - it) : it;
    const int p = it & 1;

    __syncthreads(); // A: publishes h_{t-1} (+ x buffer p on the TDM path)

    const _Float16* xbuf;
#ifdef USE_TDM
    // kick DMA for x_{t+1} into the other buffer; overlaps with this step
    if (wave == 0 && (it + 1) < LL) {
      const int tn = dir ? (t - 1) : (t + 1);
      tdm_issue_x(xin + (size_t)tn * BB * KIN,
                  (unsigned)((BB * HH + (1 - p) * BB * KIN) * sizeof(_Float16)),
                  KIN);
    }
    xbuf = p ? Xls1 : Xls0;
#else
    { // fallback: cooperative synchronous copy of x_t into buffer 0
      const _Float16* xrow = xin + (size_t)t * BB * KIN;
      for (int i = tid; i < BB * KIN / 8; i += 1024)
        *(v8h*)(Xls0 + i * 8) = *(const v8h*)(xrow + i * 8);
    }
    __syncthreads();
    xbuf = Xls0;
#endif

    v8f acc[2][4];
#pragma unroll
    for (int c = 0; c < 2; ++c)
#pragma unroll
      for (int g = 0; g < 4; ++g) {
        v8f a;
#pragma unroll
        for (int i = 0; i < 8; ++i) a[i] = bval[c][g];
        acc[c][g] = a;
      }

#pragma unroll
    for (int kc = 0; kc < KC; ++kc) {
      v16h afrag[2];
#pragma unroll
      for (int c = 0; c < 2; ++c) {
        const _Float16* src =
            (kc < KCH)
                ? (Hls + (mm[c] * 16 + n15) * HH + kc * 32 + sel * 8)
                : (xbuf + (mm[c] * 16 + n15) * KIN + (kc - KCH) * 32 + sel * 8);
        afrag[c] = load_a_frag(src);
      }
#pragma unroll
      for (int c = 0; c < 2; ++c) {
#pragma unroll
        for (int g = 0; g < 4; ++g) {
          const v16h bfrag = *(const v16h*)(wpack +
              ((size_t)((g * 16 + jj[c]) * KC + kc) * 32 + lane) * 16);
          acc[c][g] = __builtin_amdgcn_wmma_f32_16x16x32_f16(
              false, afrag[c], false, bfrag, (short)0, acc[c][g], false, false);
        }
      }
    }

    // gate combine: i, f, g, o (same (row, col) element in each accumulator)
    float hval[2][8];
#pragma unroll
    for (int c = 0; c < 2; ++c) {
#pragma unroll
      for (int i = 0; i < 8; ++i) {
        float zi = acc[c][0][i], zf = acc[c][1][i];
        float zg = acc[c][2][i], zo = acc[c][3][i];
        float si = 1.f / (1.f + __expf(-zi));
        float sf = 1.f / (1.f + __expf(-zf));
        float so = 1.f / (1.f + __expf(-zo));
        float tg = tanhf(zg);
        float cc = sf * cst[c][i] + si * tg;
        cst[c][i] = cc;
        hval[c][i] = so * tanhf(cc);
      }
    }

    __syncthreads(); // C: all reads of old h done before overwrite

#pragma unroll
    for (int c = 0; c < 2; ++c) {
#pragma unroll
      for (int i = 0; i < 8; ++i) {
        int row = mm[c] * 16 + sel * 8 + i;
        int col = jj[c] * 16 + n15;
        _Float16 hv = (_Float16)hval[c][i];
        Hls[row * HH + col] = hv;
        hout[((size_t)t * BB + row) * (2 * HH) + col_off + col] = hv;
      }
    }

#ifdef USE_TDM
    // drain the prefetch so buffer (1-p) is complete before next barrier A
    if (wave == 0) __builtin_amdgcn_s_wait_tensorcnt(0);
#endif
  }
}

// ---------------------------------------------------------------------------
// Classifier: tag_scores = h1 @ cls_w^T + cls_b. M = L*B = 32768, K = 512,
// N = 9 padded to 16. One WMMA m-tile per wave.
// ---------------------------------------------------------------------------
__global__ __launch_bounds__(256)
void cls_gemm(const _Float16* __restrict__ h1,    // (32768, 512) f16
              const _Float16* __restrict__ wcls,  // packed NT=1, KC=16
              const float* __restrict__ cls_b,    // 9
              float* __restrict__ tag,            // (32768, 9) f32
              float* __restrict__ preds) {        // (64, 512, 9) f32
  const int lane = threadIdx.x & 31;
  const int wave = threadIdx.x >> 5;
  const int mtile = blockIdx.x * 8 + wave; // 0..2047
  const int n15 = lane & 15;
  const int sel = lane >> 4;

  float biasv = (n15 < TT) ? cls_b[n15] : 0.f;
  v8f acc;
#pragma unroll
  for (int i = 0; i < 8; ++i) acc[i] = biasv;

  const _Float16* abase = h1 + (size_t)mtile * 16 * 512;
#pragma unroll
  for (int kc = 0; kc < 16; ++kc) {
    v16h a = load_a_frag(abase + (size_t)n15 * 512 + kc * 32 + sel * 8);
    v16h b = *(const v16h*)(wcls + ((size_t)(kc * 32 + lane)) * 16);
    acc = __builtin_amdgcn_wmma_f32_16x16x32_f16(
        false, a, false, b, (short)0, acc, false, false);
  }

  if (n15 < TT) {
#pragma unroll
    for (int i = 0; i < 8; ++i) {
      int row = mtile * 16 + sel * 8 + i; // row = t*64 + b
      float v = acc[i];
      tag[(size_t)row * TT + n15] = v;
      int t = row >> 6, b = row & 63;
      preds[((size_t)b * LL + t) * TT + n15] = v;
    }
  }
}

// ---------------------------------------------------------------------------
// CRF normalizer (forward algorithm). Grid = 64 (one block per batch col).
// ---------------------------------------------------------------------------
__global__ void crf_normalizer(const float* __restrict__ tag,   // (L*B, 9)
                               const int* __restrict__ amask,   // (B, L)
                               const float* __restrict__ start,
                               const float* __restrict__ endt,
                               const float* __restrict__ trans, // 9x9
                               float* __restrict__ normout) {   // 64
  int b = blockIdx.x;
  int j = threadIdx.x; // 32 threads
  __shared__ float sc[TT];
  __shared__ float tr[TT * TT];
  for (int i = j; i < TT * TT; i += 32) tr[i] = trans[i];
  if (j < TT) sc[j] = start[j] + tag[(size_t)b * TT + j];
  __syncthreads();
  for (int t = 1; t < LL; ++t) {
    float nxt = 0.f;
    if (j < TT) {
      float m = -1e30f;
      for (int i = 0; i < TT; ++i) m = fmaxf(m, sc[i] + tr[i * TT + j]);
      float s = 0.f;
      for (int i = 0; i < TT; ++i) s += __expf(sc[i] + tr[i * TT + j] - m);
      nxt = m + __logf(s) + tag[((size_t)t * BB + b) * TT + j];
    }
    int mt = amask[b * LL + t];
    __syncthreads();
    if (j < TT && mt) sc[j] = nxt;
    __syncthreads();
  }
  if (j == 0) {
    float m = -1e30f;
    for (int i = 0; i < TT; ++i) m = fmaxf(m, sc[i] + endt[i]);
    float s = 0.f;
    for (int i = 0; i < TT; ++i) s += __expf(sc[i] + endt[i] - m);
    normout[b] = m + __logf(s);
  }
}

// ---------------------------------------------------------------------------
// CRF numerator + final loss. One block of 64 threads (one per batch col).
// ---------------------------------------------------------------------------
__global__ void crf_loss(const float* __restrict__ tag,     // (L*B, 9)
                         const int* __restrict__ labels,    // (B, L)
                         const int* __restrict__ amask,     // (B, L)
                         const float* __restrict__ start,
                         const float* __restrict__ endt,
                         const float* __restrict__ trans,
                         const float* __restrict__ normv,   // 64
                         float* __restrict__ lossout) {
  int b = threadIdx.x;
  __shared__ float red[BB];
  int tg0 = labels[b * LL];
  float score = start[tg0] + tag[(size_t)b * TT + tg0];
  int prev = tg0;
  int cnt = amask[b * LL] ? 1 : 0;
  for (int t = 1; t < LL; ++t) {
    int tgt = labels[b * LL + t];
    if (amask[b * LL + t]) {
      score += trans[prev * TT + tgt] + tag[((size_t)t * BB + b) * TT + tgt];
      cnt++;
    }
    prev = tgt;
  }
  int seq_end = cnt - 1;
  score += endt[labels[b * LL + seq_end]];
  red[b] = score - normv[b];
  __syncthreads();
  if (b == 0) {
    float s = 0.f;
    for (int i = 0; i < BB; ++i) s += red[i];
    *lossout = -s;
  }
}

// ---------------------------------------------------------------------------
extern "C" void kernel_launch(void* const* d_in, const int* in_sizes, int n_in,
                              void* d_out, int out_size, void* d_ws, size_t ws_size,
                              hipStream_t stream) {
  const int*   input_ids = (const int*)d_in[0];
  const int*   labels    = (const int*)d_in[1];
  const int*   amask     = (const int*)d_in[2];
  const float* emb       = (const float*)d_in[3];
  const float* w_ih_l0f  = (const float*)d_in[4];
  const float* w_hh_l0f  = (const float*)d_in[5];
  const float* b_l0f     = (const float*)d_in[6];
  const float* w_ih_l0b  = (const float*)d_in[7];
  const float* w_hh_l0b  = (const float*)d_in[8];
  const float* b_l0b     = (const float*)d_in[9];
  const float* w_ih_l1f  = (const float*)d_in[10];
  const float* w_hh_l1f  = (const float*)d_in[11];
  const float* b_l1f     = (const float*)d_in[12];
  const float* w_ih_l1b  = (const float*)d_in[13];
  const float* w_hh_l1b  = (const float*)d_in[14];
  const float* b_l1b     = (const float*)d_in[15];
  const float* cls_w     = (const float*)d_in[16];
  const float* cls_b     = (const float*)d_in[17];
  const float* start_t   = (const float*)d_in[18];
  const float* end_t     = (const float*)d_in[19];
  const float* trans     = (const float*)d_in[20];

  float* preds = (float*)d_out;                  // (B, L, 9)
  float* loss  = (float*)d_out + (size_t)BB * LL * TT;

  // workspace carve-up (256B aligned)
  char* base = (char*)d_ws;
  size_t off = 0;
  auto carve = [&](size_t bytes) {
    char* p = base + off;
    off = (off + bytes + 255) & ~(size_t)255;
    return p;
  };
  _Float16* x_f16   = (_Float16*)carve((size_t)LL * BB * EE * 2);        // 8 MB
  _Float16* h0_f16  = (_Float16*)carve((size_t)LL * BB * 2 * HH * 2);    // 32 MB
  _Float16* h1_f16  = (_Float16*)carve((size_t)LL * BB * 2 * HH * 2);    // 32 MB
  _Float16* wp_l0f  = (_Float16*)carve((size_t)1024 * 384 * 2);
  _Float16* wp_l0b  = (_Float16*)carve((size_t)1024 * 384 * 2);
  _Float16* wp_l1f  = (_Float16*)carve((size_t)1024 * 768 * 2);
  _Float16* wp_l1b  = (_Float16*)carve((size_t)1024 * 768 * 2);
  _Float16* wp_cls  = (_Float16*)carve((size_t)16 * 512 * 2);
  float*    tag     = (float*)carve((size_t)LL * BB * TT * 4);           // 1.1 MB
  float*    normv   = (float*)carve((size_t)BB * 4);

  // 1) pack weights into WMMA-B f16 layout; A K-space = [h (256) | x (KIN)]
  {
    int tot0 = 1024 * 384;  // KC = 12
    pack_weights<<<(tot0 + 255) / 256, 256, 0, stream>>>(w_hh_l0f, HH, w_ih_l0f, EE, 1024, 12, wp_l0f, tot0);
    pack_weights<<<(tot0 + 255) / 256, 256, 0, stream>>>(w_hh_l0b, HH, w_ih_l0b, EE, 1024, 12, wp_l0b, tot0);
    int tot1 = 1024 * 768;  // KC = 24
    pack_weights<<<(tot1 + 255) / 256, 256, 0, stream>>>(w_hh_l1f, HH, w_ih_l1f, 2 * HH, 1024, 24, wp_l1f, tot1);
    pack_weights<<<(tot1 + 255) / 256, 256, 0, stream>>>(w_hh_l1b, HH, w_ih_l1b, 2 * HH, 1024, 24, wp_l1b, tot1);
    int totc = 16 * 512;    // NT = 1, KC = 16, rows 9..15 zero-padded
    pack_weights<<<(totc + 255) / 256, 256, 0, stream>>>(cls_w, 2 * HH, cls_w, 2 * HH, TT, 16, wp_cls, totc);
  }

  // 2) embedding gather (+f32->f16, (B,L) -> (L,B))
  {
    int tot = LL * BB * (EE / 4);
    embed_gather<<<(tot + 255) / 256, 256, 0, stream>>>(input_ids, emb, x_f16);
  }

  // 3) persistent bidirectional LSTM layers (2 blocks each: fwd + bwd)
  //    dynamic LDS: Hls (32 KB) + double-buffered x_t slab
  size_t sh0 = (size_t)(BB * HH + 2 * BB * EE) * sizeof(_Float16);       // 64 KB
  size_t sh1 = (size_t)(BB * HH + 2 * BB * 2 * HH) * sizeof(_Float16);   // 160 KB
  (void)hipFuncSetAttribute(reinterpret_cast<const void*>(&lstm_bidir<EE>),
                            hipFuncAttributeMaxDynamicSharedMemorySize, (int)sh0);
  (void)hipFuncSetAttribute(reinterpret_cast<const void*>(&lstm_bidir<2 * HH>),
                            hipFuncAttributeMaxDynamicSharedMemorySize, (int)sh1);
  lstm_bidir<EE><<<2, 1024, sh0, stream>>>(x_f16, wp_l0f, wp_l0b, b_l0f, b_l0b, h0_f16);
  lstm_bidir<2 * HH><<<2, 1024, sh1, stream>>>(h0_f16, wp_l1f, wp_l1b, b_l1f, b_l1b, h1_f16);

  // 4) classifier GEMM (writes tag scores + preds output)
  cls_gemm<<<256, 256, 0, stream>>>(h1_f16, wp_cls, cls_b, tag, preds);

  // 5) CRF
  crf_normalizer<<<BB, 32, 0, stream>>>(tag, amask, start_t, end_t, trans, normv);
  crf_loss<<<1, BB, 0, stream>>>(tag, labels, amask, start_t, end_t, trans, normv, loss);

  (void)in_sizes; (void)n_in; (void)out_size; (void)ws_size;
}
